// QuantumAttentionMixingCore_83863531421715
// MI455X (gfx1250) — matrix operand order repaired and checked by reading, not compiled, not run on hardware
//
#include <hip/hip_runtime.h>
#include <hip/hip_fp16.h>

typedef __attribute__((ext_vector_type(16))) _Float16 v16h;
typedef __attribute__((ext_vector_type(8)))  float    v8f;

#define NQ      6
#define SDIM    64
#define BATCH   4096
#define TCH     8
#define BT      (BATCH * TCH)      // 32768
#define DDIM    1024
#define NCP     60                 // circuit params per (b,t): 2 layers * 30
#define NPL     30                 // params per layer
#define PSTRIDE 64                 // padded param row stride

// LDS offset of a generic pointer to __shared__ (low 32 bits of LDS-aperture addr)
__device__ __forceinline__ unsigned lds_off(const void* p) {
    return (unsigned)(unsigned long long)p;
}

// =====================================================================
// Kernel 1: params = sigmoid(x @ W^T + b)   via v_wmma_f32_16x16x32_f16
//   x: (32768, 1024) f32 ; W: (60, 1024) f32 ; params: (32768, 64) f32
// Block: 256 threads (8 waves); 128 rows/block. Wave w: rows 16w..16w+15,
// 4 N-tiles (64 padded cols). K: 32 chunks of 32.
// A tile: double-buffered f32 in LDS, filled by global_load_async_to_lds_b128
// (ASYNCcnt) one chunk ahead of compute; converted to f16 at fragment gather.
// B tile: double-buffered f16 (VALU-staged, zero-padded past col 59).
// =====================================================================
__global__ __launch_bounds__(256) void gemm_sigmoid_kernel(
    const float* __restrict__ x, const float* __restrict__ w,
    const float* __restrict__ bias, float* __restrict__ params)
{
    // row stride 36 floats = 144B: 16B-aligned rows for B128 async writes,
    // and 36*l mod 64 spreads banks across the 16 gather lanes.
    __shared__ __align__(16) float    ldsAf[2][128][36];  // 36 KiB
    __shared__            unsigned    ldsB[2][64][17];    // 8.5 KiB (half2-packed)

    const int tid  = threadIdx.x;
    const int lane = tid & 31;
    const int wv   = tid >> 5;          // 0..7
    const int rowBlock = blockIdx.x * 128;
    const bool lo  = lane < 16;
    const int l15  = lane & 15;

    v8f acc[4];
    #pragma unroll
    for (int nt = 0; nt < 4; ++nt)
        #pragma unroll
        for (int e = 0; e < 8; ++e) acc[nt][e] = 0.0f;

    // ---- helper lambdas -------------------------------------------------
    auto issueA = [&](int buf, int k0) {
        // 128 rows x 32 f32 = 1024 x 16B chunks; 4 per thread, async -> LDS
        #pragma unroll
        for (int i = 0; i < 4; ++i) {
            const int e = tid + i * 256;
            const int r = e >> 3, c = e & 7;          // r:0..127, c:0..7 (16B chunks)
            const float* gp = x + (size_t)(rowBlock + r) * DDIM + k0 + 4 * c;
            const unsigned ld = lds_off(&ldsAf[buf][r][4 * c]);
            asm volatile("global_load_async_to_lds_b128 %0, %1, off"
                         :: "v"(ld), "v"((unsigned long long)gp) : "memory");
        }
    };
    auto stageB = [&](int buf, int k0) {
        // 64 cols x 16 half2 = 1024 entries; 4 per thread
        #pragma unroll
        for (int i = 0; i < 4; ++i) {
            const int e  = tid + i * 256;
            const int pc = e >> 4;                    // padded col 0..63
            const int kp = e & 15;
            union { _Float16 h[2]; unsigned u; } pk;
            if (pc < NCP) {
                const float2 v = *(const float2*)(w + (size_t)pc * DDIM + k0 + 2 * kp);
                pk.h[0] = (_Float16)v.x; pk.h[1] = (_Float16)v.y;
            } else {
                pk.h[0] = (_Float16)0.0f; pk.h[1] = (_Float16)0.0f;
            }
            ldsB[buf][pc][kp] = pk.u;
        }
    };

    // ---- prologue: chunk 0 in flight ----
    stageB(0, 0);
    issueA(0, 0);

    for (int kt = 0; kt < 32; ++kt) {
        const int cur = kt & 1;
        asm volatile("s_wait_asynccnt 0x0" ::: "memory");  // my A(kt) copies landed
        __syncthreads();                                   // everyone's A+B visible; prev compute done

        if (kt < 31) {                                     // overlap next copies with compute
            issueA(cur ^ 1, (kt + 1) * 32);
            stageB(cur ^ 1, (kt + 1) * 32);
        }

        // ---- A fragment (16x32 f16): lanes 0-15: K{0..7,16..23}; 16-31: K{8..15,24..31}
        union { unsigned u[8]; v16h h; } af;
        const float* arow = &ldsAf[cur][wv * 16 + l15][0];
        const int kb = lo ? 0 : 8;
        #pragma unroll
        for (int j = 0; j < 8; ++j) {
            const int k = kb + 2 * (j & 3) + ((j >> 2) << 4);
            const float2 d = *(const float2*)(arow + k);
            union { _Float16 h[2]; unsigned u; } pk;
            pk.h[0] = (_Float16)d.x; pk.h[1] = (_Float16)d.y;
            af.u[j] = pk.u;
        }

        // ---- 4 N-tiles: B fragment (32x16 f16): lanes 0-15: K{0..7,16..23}; 16-31: K{8..15,24..31}
        const int bbase = lo ? 0 : 4;
        #pragma unroll
        for (int nt = 0; nt < 4; ++nt) {
            union { unsigned u[8]; v16h h; } bf;
            #pragma unroll
            for (int j = 0; j < 8; ++j) bf.u[j] = ldsB[cur][nt * 16 + l15][bbase + (j & 3) + ((j >> 2) << 3)];
            acc[nt] = __builtin_amdgcn_wmma_f32_16x16x32_f16(
                false, af.h, false, bf.h, (short)0, acc[nt], false, false);
        }
    }

    // ---- bias + sigmoid + store (C layout: VGPR r -> M=r (lanes<16) / M=8+r) ----
    #pragma unroll
    for (int nt = 0; nt < 4; ++nt) {
        const int col = nt * 16 + l15;
        const float bv = (col < NCP) ? bias[col] : 0.0f;
        #pragma unroll
        for (int r = 0; r < 8; ++r) {
            const int row = rowBlock + wv * 16 + r + (lo ? 0 : 8);
            const float v = acc[nt][r] + bv;
            params[(size_t)row * PSTRIDE + col] = 1.0f / (1.0f + __expf(-v));
        }
    }
}

// =====================================================================
// Quantum circuit helpers: 64-amplitude state held entirely in VGPRs.
// wire w acts on bit (5-w) of the flat index (wire 0 = MSB).
// All call sites have compile-time wires -> loops fully constant-folded.
// =====================================================================
__device__ __forceinline__ void apply_rx(float* re, float* im, int wire, float th) {
    const float c = __cosf(0.5f * th), s = __sinf(0.5f * th);
    const int st = 1 << (5 - wire);
    #pragma unroll
    for (int a = 0; a < 64; ++a) {
        if (a & st) continue;
        const int b = a + st;
        const float r0 = re[a], i0 = im[a], r1 = re[b], i1 = im[b];
        re[a] =  c * r0 + s * i1;  im[a] =  c * i0 - s * r1;
        re[b] =  s * i0 + c * r1;  im[b] = -s * r0 + c * i1;
    }
}

__device__ __forceinline__ void apply_ry(float* re, float* im, int wire, float th) {
    const float c = __cosf(0.5f * th), s = __sinf(0.5f * th);
    const int st = 1 << (5 - wire);
    #pragma unroll
    for (int a = 0; a < 64; ++a) {
        if (a & st) continue;
        const int b = a + st;
        const float r0 = re[a], i0 = im[a], r1 = re[b], i1 = im[b];
        re[a] = c * r0 - s * r1;  im[a] = c * i0 - s * i1;
        re[b] = s * r0 + c * r1;  im[b] = s * i0 + c * i1;
    }
}

__device__ __forceinline__ void apply_rz(float* re, float* im, int wire, float th) {
    const float c = __cosf(0.5f * th), s = __sinf(0.5f * th);
    const int st = 1 << (5 - wire);
    #pragma unroll
    for (int a = 0; a < 64; ++a) {
        const float sg = (a & st) ? s : -s;
        const float r = re[a], i0 = im[a];
        re[a] = c * r  - sg * i0;
        im[a] = c * i0 + sg * r;
    }
}

__device__ __forceinline__ void apply_crx(float* re, float* im, int cw, int tw, float th) {
    const float c = __cosf(0.5f * th), s = __sinf(0.5f * th);
    const int cm = 1 << (5 - cw), st = 1 << (5 - tw);
    #pragma unroll
    for (int a = 0; a < 64; ++a) {
        if (!(a & cm) || (a & st)) continue;
        const int b = a + st;
        const float r0 = re[a], i0 = im[a], r1 = re[b], i1 = im[b];
        re[a] =  c * r0 + s * i1;  im[a] =  c * i0 - s * r1;
        re[b] =  s * i0 + c * r1;  im[b] = -s * r0 + c * i1;
    }
}

__device__ __forceinline__ void ansatz_layer(float* re, float* im,
                                             const float* __restrict__ p, int off) {
    int idx = off;
    #pragma unroll
    for (int q = 0; q < NQ; ++q) {
        apply_rx(re, im, q, p[idx]);
        apply_ry(re, im, q, p[idx + 1]);
        apply_rz(re, im, q, p[idx + 2]);
        idx += 3;
    }
    #pragma unroll
    for (int q = 0; q < NQ; ++q) { apply_crx(re, im, q, (q + 1) % NQ, p[idx]); ++idx; }
    #pragma unroll
    for (int q = NQ - 1; q >= 0; --q) { apply_crx(re, im, q, (q + 5) % NQ, p[idx]); ++idx; }
}

// =====================================================================
// Kernel 2: per-(b,t) circuit, 2 layers, from |0..0>. One thread per state.
// =====================================================================
__global__ __launch_bounds__(256) void circuit_kernel(
    const float* __restrict__ params, float2* __restrict__ evolved)
{
    const int bt = blockIdx.x * blockDim.x + threadIdx.x;   // 0..32767
    float re[64], im[64];
    #pragma unroll
    for (int a = 0; a < 64; ++a) { re[a] = 0.0f; im[a] = 0.0f; }
    re[0] = 1.0f;

    const float* p = params + (size_t)bt * PSTRIDE;
    ansatz_layer(re, im, p, 0);
    ansatz_layer(re, im, p, NPL);

    float2* out = evolved + (size_t)bt * SDIM;
    #pragma unroll
    for (int a = 0; a < 64; ++a) out[a] = make_float2(re[a], im[a]);
}

// =====================================================================
// Kernel 3: LCU mix over chunks + L2 normalize + QFF layer + <X,Y,Z> feats.
// One thread per batch element b.
// =====================================================================
__global__ __launch_bounds__(256) void mix_qff_kernel(
    const float2* __restrict__ evolved,
    const float* __restrict__ mre, const float* __restrict__ mim,
    const float* __restrict__ qff, float* __restrict__ qfeat)
{
    const int b = blockIdx.x * blockDim.x + threadIdx.x;    // 0..4095

    float cr[TCH], ci[TCH];
    float denom = 1e-8f;
    #pragma unroll
    for (int t = 0; t < TCH; ++t) {
        cr[t] = mre[t]; ci[t] = mim[t];
        denom += __fsqrt_rn(cr[t] * cr[t] + ci[t] * ci[t]);
    }
    const float inv = 1.0f / denom;
    #pragma unroll
    for (int t = 0; t < TCH; ++t) { cr[t] *= inv; ci[t] *= inv; }

    float re[64], im[64];
    #pragma unroll
    for (int a = 0; a < 64; ++a) { re[a] = 0.0f; im[a] = 0.0f; }
    #pragma unroll
    for (int t = 0; t < TCH; ++t) {
        const float2* ev = evolved + ((size_t)b * TCH + t) * SDIM;
        #pragma unroll
        for (int a = 0; a < 64; ++a) {
            const float2 e = ev[a];
            re[a] += e.x * cr[t] - e.y * ci[t];
            im[a] += e.x * ci[t] + e.y * cr[t];
        }
    }
    float nrm = 0.0f;
    #pragma unroll
    for (int a = 0; a < 64; ++a) nrm += re[a] * re[a] + im[a] * im[a];
    const float sc = 1.0f / (__fsqrt_rn(nrm) + 1e-9f);
    #pragma unroll
    for (int a = 0; a < 64; ++a) { re[a] *= sc; im[a] *= sc; }

    ansatz_layer(re, im, qff, 0);   // shared 30-param layer

    // expectations <X_w>, <Y_w>, <Z_w>
    float* fo = qfeat + (size_t)b * 18;
    #pragma unroll
    for (int wq = 0; wq < NQ; ++wq) {
        const int st = 1 << (5 - wq);
        float xr = 0.0f, xi = 0.0f, zz = 0.0f;
        #pragma unroll
        for (int a = 0; a < 64; ++a) {
            if (a & st) continue;
            const int bb = a + st;
            xr += re[a] * re[bb] + im[a] * im[bb];   // Re(conj(a)*b)
            xi += re[a] * im[bb] - im[a] * re[bb];   // Im(conj(a)*b)
            zz += (re[a] * re[a] + im[a] * im[a]) - (re[bb] * re[bb] + im[bb] * im[bb]);
        }
        fo[wq]      = 2.0f * xr;
        fo[6 + wq]  = 2.0f * xi;
        fo[12 + wq] = zz;
    }
}

// =====================================================================
// Kernel 4: out = qfeat @ Wout^T + bout  (4096x18 @ 18x1024). Store-bound.
// =====================================================================
__global__ __launch_bounds__(256) void outproj_kernel(
    const float* __restrict__ qfeat, const float* __restrict__ wout,
    const float* __restrict__ bout, float* __restrict__ out)
{
    __shared__ float f[18];
    const int b = blockIdx.x;
    if (threadIdx.x < 18) f[threadIdx.x] = qfeat[(size_t)b * 18 + threadIdx.x];
    __syncthreads();
    for (int d = threadIdx.x; d < DDIM; d += blockDim.x) {
        float acc = bout[d];
        const float* wr = wout + (size_t)d * 18;
        #pragma unroll
        for (int j = 0; j < 18; ++j) acc += f[j] * wr[j];
        out[(size_t)b * DDIM + d] = acc;
    }
}

// =====================================================================
extern "C" void kernel_launch(void* const* d_in, const int* in_sizes, int n_in,
                              void* d_out, int out_size, void* d_ws, size_t ws_size,
                              hipStream_t stream)
{
    const float* x   = (const float*)d_in[0];   // (4096, 8, 1024)
    const float* ppw = (const float*)d_in[1];   // (60, 1024)
    const float* ppb = (const float*)d_in[2];   // (60,)
    const float* mre = (const float*)d_in[3];   // (8,)
    const float* mim = (const float*)d_in[4];   // (8,)
    const float* qff = (const float*)d_in[5];   // (30,)
    const float* opw = (const float*)d_in[6];   // (1024, 18)
    const float* opb = (const float*)d_in[7];   // (1024,)
    float* out = (float*)d_out;                 // (4096, 1024)

    char* ws = (char*)d_ws;
    float*  params  = (float*)ws;                                        //  8 MiB
    float2* evolved = (float2*)(ws + (size_t)BT * PSTRIDE * 4);          // 16 MiB
    float*  qfeat   = (float*)(ws + (size_t)BT * PSTRIDE * 4
                                  + (size_t)BT * SDIM * 8);              // 288 KiB

    hipLaunchKernelGGL(gemm_sigmoid_kernel, dim3(BT / 128), dim3(256), 0, stream,
                       x, ppw, ppb, params);
    hipLaunchKernelGGL(circuit_kernel, dim3(BT / 256), dim3(256), 0, stream,
                       params, evolved);
    hipLaunchKernelGGL(mix_qff_kernel, dim3(BATCH / 256), dim3(256), 0, stream,
                       evolved, mre, mim, qff, qfeat);
    hipLaunchKernelGGL(outproj_kernel, dim3(BATCH), dim3(256), 0, stream,
                       qfeat, opw, opb, out);
}
